// SingleShiftedWindow_36558761623690
// MI455X (gfx1250) — compile-verified, matrix operand order checked by reading, not run on hardware
//
#include <hip/hip_runtime.h>

typedef _Float16 v16h __attribute__((ext_vector_type(16)));
typedef float    v8f  __attribute__((ext_vector_type(8)));

static constexpr int DIMC  = 256;
static constexpr int QS    = 56;
static constexpr int HD    = 32;     // head dim
static constexpr int LPAD  = 64;     // 49 -> 64
static constexpr int BATCH = 16;
static constexpr int NWIN  = 64;     // 8x8 windows
static constexpr int FFW   = 1024;
static constexpr int NH    = 8;
static constexpr int MWIN  = BATCH * NWIN * LPAD;   // 65536 padded window rows
static constexpr int MTOK  = BATCH * QS * QS;       // 50176 tokens
static constexpr float SCALE = 0.17677669529663687f; // 1/sqrt(32)

// ---------------- WMMA helpers (gfx1250, wave32) ----------------

__device__ inline v8f wmma16(v16h a, v16h b, v8f c) {
  return __builtin_amdgcn_wmma_f32_16x16x32_f16(false, a, false, b, (short)0, c,
                                                false, false);
}

// A fragment 16x32 (row-major source, ld in elements). Per ISA 7.12.2:
// lane = m%16, K split 0-7/16-23 (lanes 0-15) vs 8-15/24-31 (lanes 16-31).
// Contiguous 8-element chunks per lane -> merges into (ds_)load_b128.
// Also used for B fragments whenever B is stored K-major (B^T row-major):
// layouts are symmetric with n taking the role of m.
__device__ inline v16h frag_a(const _Float16* p, int ld) {
  const int lane = threadIdx.x & 31;
  const int m  = lane & 15;
  const int kb = (lane >> 4) * 8;
  const _Float16* r = p + m * ld + kb;
  v16h a;
#pragma unroll
  for (int j = 0; j < 4; ++j) {
    a[2 * j]     = r[2 * j];
    a[2 * j + 1] = r[2 * j + 1];
    a[8 + 2 * j] = r[16 + 2 * j];
    a[9 + 2 * j] = r[16 + 2 * j + 1];
  }
  return a;
}

__device__ inline void store_d_f32(float* p, int ld, v8f d) {
  const int lane = threadIdx.x & 31;
  const int n  = lane & 15;
  const int mb = (lane >> 4) * 8;
#pragma unroll
  for (int i = 0; i < 8; ++i) p[(mb + i) * ld + n] = d[i];
}

__device__ inline void store_d_f16(_Float16* p, int ld, v8f d) {
  const int lane = threadIdx.x & 31;
  const int n  = lane & 15;
  const int mb = (lane >> 4) * 8;
#pragma unroll
  for (int i = 0; i < 8; ++i) p[(mb + i) * ld + n] = (_Float16)d[i];
}

__device__ inline float wave_red_add(float v) {
#pragma unroll
  for (int o = 16; o > 0; o >>= 1) v += __shfl_xor(v, o, 32);
  return v;
}

// ---------------- Async-LDS staged 128x64 GEMM core ----------------
// A: f16 [M x K] row-major (lda); Bt: f16 transposed weights [N x K] row-major
// (ldk = K stride), so both A and B tiles are K-major => all fragment reads
// are contiguous b128 chunks. 256 threads = 8 waves; wave (wm,wn) owns a
// 32x32 output tile (2x2 WMMA). Tiles staged with
// GLOBAL_LOAD_ASYNC_TO_LDS_B128 (ASYNCcnt), 2-stage double buffer.
static constexpr int SA_LD = 48;  // 96B rows: 16B-aligned chunks
static constexpr int SB_LD = 48;

__device__ inline void async_ld16(uint32_t lds_addr, int voff_bytes,
                                  unsigned long long saddr) {
  asm volatile("global_load_async_to_lds_b128 %0, %1, %2"
               :: "v"(lds_addr), "v"(voff_bytes), "s"(saddr)
               : "memory");
}

template <int K>
__device__ inline void gemm_core(const _Float16* Ablk, int lda,
                                 const _Float16* Bt, int ldk, int ncol0,
                                 _Float16* sA, _Float16* sB, v8f acc[2][2]) {
  const int tid  = threadIdx.x;
  const int wave = tid >> 5;
  const int wm   = wave & 3;
  const int wn   = wave >> 2;
  constexpr int NT = K / 32;

  // Per-thread 16B chunk coordinates (each asm stmt = 1 wave-instruction;
  // 3 async ops per wave per tile: 2 for A (128x32), 1 for B (64x32)).
  const int arow0 = tid >> 2;                 // 0..63
  const int arow1 = arow0 + 64;               // 64..127
  const int ac    = (tid & 3) * 8;
  const int brow  = tid >> 2;                 // 0..63 (n-local)
  const int bc    = (tid & 3) * 8;
  const unsigned long long aBase = (unsigned long long)(uintptr_t)Ablk;
  const unsigned long long bBase = (unsigned long long)(uintptr_t)Bt;
  const uint32_t la0 = (uint32_t)(uintptr_t)(sA + arow0 * SA_LD + ac);
  const uint32_t la1 = (uint32_t)(uintptr_t)(sA + arow1 * SA_LD + ac);
  const uint32_t lb  = (uint32_t)(uintptr_t)(sB + brow * SB_LD + bc);
  const int aoff0 = (arow0 * lda + ac) * 2;
  const int aoff1 = (arow1 * lda + ac) * 2;
  const int boff  = ((ncol0 + brow) * ldk + bc) * 2;
  constexpr uint32_t ABUF = 128 * SA_LD * 2;  // bytes per A buffer
  constexpr uint32_t BBUF = 64 * SB_LD * 2;   // bytes per B buffer

  auto issue = [&](int buf, int kt) {
    async_ld16(la0 + buf * ABUF, aoff0 + kt * 2, aBase);
    async_ld16(la1 + buf * ABUF, aoff1 + kt * 2, aBase);
    async_ld16(lb + buf * BBUF, boff + kt * 2, bBase);
  };

  issue(0, 0);
  issue(1, 32);
#pragma unroll 1
  for (int i = 0; i < NT; ++i) {
    if (i + 1 < NT)
      asm volatile("s_wait_asynccnt 3" ::: "memory");
    else
      asm volatile("s_wait_asynccnt 0" ::: "memory");
    __syncthreads();
    const _Float16* cA = sA + (i & 1) * (128 * SA_LD);
    const _Float16* cB = sB + (i & 1) * (64 * SB_LD);
    v16h a0 = frag_a(cA + (wm * 32) * SA_LD, SA_LD);
    v16h a1 = frag_a(cA + (wm * 32 + 16) * SA_LD, SA_LD);
    v16h b0 = frag_a(cB + (wn * 32) * SB_LD, SB_LD);       // K-major B
    v16h b1 = frag_a(cB + (wn * 32 + 16) * SB_LD, SB_LD);
    acc[0][0] = wmma16(a0, b0, acc[0][0]);
    acc[0][1] = wmma16(a0, b1, acc[0][1]);
    acc[1][0] = wmma16(a1, b0, acc[1][0]);
    acc[1][1] = wmma16(a1, b1, acc[1][1]);
    __syncthreads();
    if (i + 2 < NT) issue(i & 1, (i + 2) * 32);
  }
}

#define GEMM_PROLOGUE()                                    \
  __shared__ _Float16 sA[2 * 128 * SA_LD];                 \
  __shared__ _Float16 sB[2 * 64 * SB_LD];                  \
  const int tid = threadIdx.x;                             \
  const int wave = tid >> 5, wm = wave & 3, wn = wave >> 2;\
  const int lane = tid & 31, n_in = lane & 15, mb = (lane >> 4) * 8; \
  const int rowblk = blockIdx.y * 128;                     \
  const int ncol0 = blockIdx.x * 64;                       \
  v8f acc[2][2];                                           \
  { v8f z = {0, 0, 0, 0, 0, 0, 0, 0};                      \
    acc[0][0] = z; acc[0][1] = z; acc[1][0] = z; acc[1][1] = z; } \
  (void)n_in; (void)mb;

// -------- Kernel 0: f32 -> f16 weight conversion WITH transpose ----------
// src: W[k][n] f32 (K x N row-major) -> dst: Wt[n][k] f16 (N x K row-major)
__global__ __launch_bounds__(256) void k_w2ht(const float* __restrict__ src,
                                              _Float16* __restrict__ dst,
                                              int Kd, int Nd) {
  int i = blockIdx.x * 256 + threadIdx.x;
  if (i < Kd * Nd) {
    int n = i / Kd, kk = i - n * Kd;
    dst[i] = (_Float16)src[kk * Nd + n];
  }
}

// ---------------- Kernel 1: LN1 + shift + window partition -> f16 ----------
__global__ __launch_bounds__(256) void k_ln1(const float* __restrict__ x,
                                             const float* __restrict__ g,
                                             const float* __restrict__ bb,
                                             _Float16* __restrict__ hwin) {
  const int wave = threadIdx.x >> 5, lane = threadIdx.x & 31;
  const int row = blockIdx.x * 8 + wave;
  const int win = row >> 6, l = row & 63;
  _Float16* dst = hwin + (size_t)row * DIMC;
  if (l >= 49) {
#pragma unroll
    for (int j = 0; j < 8; ++j) dst[lane + 32 * j] = (_Float16)0.f;
    return;
  }
  const int b_ = win >> 6, wl = win & 63, wh = wl >> 3, ww = wl & 7;
  const int r = l / 7, c = l % 7;
  const int h = (wh * 7 + r + 3) % QS;
  const int w = (ww * 7 + c + 3) % QS;
  const float* xp = x + (size_t)b_ * DIMC * QS * QS + h * QS + w;
  float vals[8], s = 0.f, s2 = 0.f;
#pragma unroll
  for (int j = 0; j < 8; ++j) {
    float t = xp[(size_t)(lane + 32 * j) * (QS * QS)];
    vals[j] = t; s += t; s2 += t * t;
  }
  s = wave_red_add(s); s2 = wave_red_add(s2);
  const float mean = s * (1.f / DIMC);
  const float var  = s2 * (1.f / DIMC) - mean * mean;
  const float rstd = rsqrtf(var + 1e-5f);
#pragma unroll
  for (int j = 0; j < 8; ++j) {
    int ch = lane + 32 * j;
    dst[ch] = (_Float16)((vals[j] - mean) * rstd * g[ch] + bb[ch]);
  }
}

// ---------------- Kernel 2: QKV GEMM (65536 x 768 x 256) ----------------
// q, k stored row-major [row][256]; v stored TRANSPOSED per window:
// vT[win][c][l] (c = head*32+d, l = 0..63) so attention P*V B-fragments are
// contiguous; the transposed store is 8 consecutive f16 per lane (b128).
__global__ __launch_bounds__(256) void k_qkv(const _Float16* __restrict__ hwin,
                                             const _Float16* __restrict__ wq16t,
                                             const float* __restrict__ qkv_b,
                                             _Float16* __restrict__ q,
                                             _Float16* __restrict__ k,
                                             _Float16* __restrict__ vt) {
  GEMM_PROLOGUE();
  gemm_core<DIMC>(hwin + (size_t)rowblk * DIMC, DIMC, wq16t, DIMC, ncol0,
                  sA, sB, acc);
#pragma unroll
  for (int mi = 0; mi < 2; ++mi)
#pragma unroll
    for (int ni = 0; ni < 2; ++ni) {
      int colBase = ncol0 + wn * 32 + ni * 16;
      int sel = colBase >> 8;
      int col = (colBase & 255) + n_in;
      float bias = qkv_b[colBase + n_in];
      if (sel < 2) {
        _Float16* dst = (sel == 0) ? q : k;
#pragma unroll
        for (int i = 0; i < 8; ++i) {
          int row = rowblk + wm * 32 + mi * 16 + mb + i;
          dst[(size_t)row * DIMC + col] = (_Float16)(acc[mi][ni][i] + bias);
        }
      } else {
        int row0 = rowblk + wm * 32 + mi * 16 + mb;  // 8-row run, win-aligned
        _Float16* dst =
            vt + (size_t)(row0 >> 6) * (DIMC * LPAD) + col * LPAD + (row0 & 63);
#pragma unroll
        for (int i = 0; i < 8; ++i)
          dst[i] = (_Float16)(acc[mi][ni][i] + bias);
      }
    }
}

// ---------------- Kernel 3: windowed attention, one block per (win, head) ---
__global__ __launch_bounds__(128) void k_attn(const _Float16* __restrict__ q,
                                              const _Float16* __restrict__ k,
                                              const _Float16* __restrict__ vt,
                                              _Float16* __restrict__ o) {
  __shared__ float    sS[64 * 64];
  __shared__ _Float16 sP[64 * 64];
  __shared__ int      sreg[64];
  const int win = blockIdx.x >> 3, head = blockIdx.x & 7;
  const int tid = threadIdx.x, wave = tid >> 5;
  if (tid < 64) {
    int wl = win & 63, wh = wl >> 3, ww = wl & 7;
    int reg = -1 - tid;
    if (tid < 49) {
      int r = tid / 7, c = tid % 7;
      int bh = (wh < 7) ? 0 : ((r < 4) ? 1 : 2);
      int bw = (ww < 7) ? 0 : ((c < 4) ? 1 : 2);
      reg = bh * 3 + bw;
    }
    sreg[tid] = reg;
  }
  const _Float16* qb = q + (size_t)win * LPAD * DIMC + head * HD;
  const _Float16* kb = k + (size_t)win * LPAD * DIMC + head * HD;
  const _Float16* vb = vt + ((size_t)win * DIMC + head * HD) * LPAD;
  const int m0 = wave * 16;
  // S = Q * K^T   (K dim = 32 = head dim: single WMMA per tile)
  v16h aQ = frag_a(qb + m0 * DIMC, DIMC);
#pragma unroll
  for (int nt = 0; nt < 4; ++nt) {
    v16h bK = frag_a(kb + nt * 16 * DIMC, DIMC);  // transposed-B == A layout
    v8f accS = {0, 0, 0, 0, 0, 0, 0, 0};
    accS = wmma16(aQ, bK, accS);
    store_d_f32(sS + m0 * 64 + nt * 16, 64, accS);
  }
  __syncthreads();
  // rowwise softmax with Swin region mask; rows >= 49 -> zero P
  if (tid < 64) {
    const int i = tid;
    float* srow = sS + i * 64;
    _Float16* prow = sP + i * 64;
    if (i < 49) {
      const int ri = sreg[i];
      float mx = -1e30f;
      for (int j = 0; j < 49; ++j) {
        float t = srow[j] * SCALE + ((sreg[j] == ri) ? 0.f : -100.f);
        srow[j] = t;
        mx = fmaxf(mx, t);
      }
      float sum = 0.f;
      for (int j = 0; j < 49; ++j) {
        float e = __expf(srow[j] - mx);
        srow[j] = e;
        sum += e;
      }
      float inv = 1.f / sum;
      for (int j = 0; j < 49; ++j) prow[j] = (_Float16)(srow[j] * inv);
      for (int j = 49; j < 64; ++j) prow[j] = (_Float16)0.f;
    } else {
      for (int j = 0; j < 64; ++j) prow[j] = (_Float16)0.f;
    }
  }
  __syncthreads();
  // O = P * V   (K dim = 64 -> two WMMA; V is K-major: contiguous loads)
#pragma unroll
  for (int dt = 0; dt < 2; ++dt) {
    v8f accO = {0, 0, 0, 0, 0, 0, 0, 0};
#pragma unroll
    for (int jc = 0; jc < 2; ++jc) {
      v16h aP = frag_a(sP + m0 * 64 + jc * 32, 64);
      v16h bV = frag_a(vb + dt * 16 * LPAD + jc * 32, LPAD);
      accO = wmma16(aP, bV, accO);
    }
    store_d_f16(o + (size_t)(win * LPAD + m0) * DIMC + head * HD + dt * 16,
                DIMC, accO);
  }
}

// ---------------- Kernel 4: proj GEMM + reverse window/roll + residual ------
__global__ __launch_bounds__(256) void k_proj(const _Float16* __restrict__ owin,
                                              const _Float16* __restrict__ wp16t,
                                              const float* __restrict__ proj_b,
                                              const float* __restrict__ x,
                                              float* __restrict__ x1) {
  GEMM_PROLOGUE();
  gemm_core<DIMC>(owin + (size_t)rowblk * DIMC, DIMC, wp16t, DIMC, ncol0,
                  sA, sB, acc);
#pragma unroll
  for (int mi = 0; mi < 2; ++mi)
#pragma unroll
    for (int ni = 0; ni < 2; ++ni) {
      int col = ncol0 + wn * 32 + ni * 16 + n_in;
      float bias = proj_b[col];
#pragma unroll
      for (int i = 0; i < 8; ++i) {
        int row = rowblk + wm * 32 + mi * 16 + mb + i;
        int l = row & 63;
        if (l >= 49) continue;
        int win = row >> 6;
        int b_ = win >> 6, wl = win & 63, wh = wl >> 3, ww = wl & 7;
        int r = l / 7, c = l % 7;
        int h = (wh * 7 + r + 3) % QS;
        int w = (ww * 7 + c + 3) % QS;
        float xv = x[((size_t)(b_ * DIMC + col) * QS + h) * QS + w];
        x1[((size_t)(b_ * QS + h) * QS + w) * DIMC + col] =
            xv + acc[mi][ni][i] + bias;
      }
    }
}

// ---------------- Kernel 5: LN2 (contiguous tokens) -> f16 ----------------
__global__ __launch_bounds__(256) void k_ln2(const float* __restrict__ x1,
                                             const float* __restrict__ g,
                                             const float* __restrict__ bb,
                                             _Float16* __restrict__ h2) {
  const int wave = threadIdx.x >> 5, lane = threadIdx.x & 31;
  const int row = blockIdx.x * 8 + wave;
  const float* xp = x1 + (size_t)row * DIMC;
  _Float16* dst = h2 + (size_t)row * DIMC;
  float vals[8], s = 0.f, s2 = 0.f;
#pragma unroll
  for (int j = 0; j < 8; ++j) {
    float t = xp[lane + 32 * j];
    vals[j] = t; s += t; s2 += t * t;
  }
  s = wave_red_add(s); s2 = wave_red_add(s2);
  const float mean = s * (1.f / DIMC);
  const float var  = s2 * (1.f / DIMC) - mean * mean;
  const float rstd = rsqrtf(var + 1e-5f);
#pragma unroll
  for (int j = 0; j < 8; ++j) {
    int ch = lane + 32 * j;
    dst[ch] = (_Float16)((vals[j] - mean) * rstd * g[ch] + bb[ch]);
  }
}

// ---------------- Kernel 6: FFN1 GEMM + bias + ReLU ----------------
__global__ __launch_bounds__(256) void k_ffn1(const _Float16* __restrict__ h2,
                                              const _Float16* __restrict__ w1t,
                                              const float* __restrict__ b1,
                                              _Float16* __restrict__ f1) {
  GEMM_PROLOGUE();
  gemm_core<DIMC>(h2 + (size_t)rowblk * DIMC, DIMC, w1t, DIMC, ncol0,
                  sA, sB, acc);
#pragma unroll
  for (int mi = 0; mi < 2; ++mi)
#pragma unroll
    for (int ni = 0; ni < 2; ++ni) {
      int col = ncol0 + wn * 32 + ni * 16 + n_in;
      float bias = b1[col];
#pragma unroll
      for (int i = 0; i < 8; ++i) {
        int row = rowblk + wm * 32 + mi * 16 + mb + i;
        f1[(size_t)row * FFW + col] =
            (_Float16)fmaxf(acc[mi][ni][i] + bias, 0.f);
      }
    }
}

// ------ Kernel 7: FFN2 GEMM + bias + layer_scale + residual + BCHW store ----
__global__ __launch_bounds__(256) void k_ffn2(const _Float16* __restrict__ f1,
                                              const _Float16* __restrict__ w2t,
                                              const float* __restrict__ b2,
                                              const float* __restrict__ ls,
                                              const float* __restrict__ x1,
                                              float* __restrict__ out) {
  GEMM_PROLOGUE();
  gemm_core<FFW>(f1 + (size_t)rowblk * FFW, FFW, w2t, FFW, ncol0, sA, sB, acc);
#pragma unroll
  for (int mi = 0; mi < 2; ++mi)
#pragma unroll
    for (int ni = 0; ni < 2; ++ni) {
      int col = ncol0 + wn * 32 + ni * 16 + n_in;
      float bias = b2[col], sc = ls[col];
#pragma unroll
      for (int i = 0; i < 8; ++i) {
        int row = rowblk + wm * 32 + mi * 16 + mb + i;
        int b_ = row / (QS * QS);
        int rem = row % (QS * QS);
        int h = rem / QS, w = rem % QS;
        float d = (acc[mi][ni][i] + bias) * sc + x1[(size_t)row * DIMC + col];
        out[((size_t)(b_ * DIMC + col) * QS + h) * QS + w] = d;
      }
    }
}

// ---------------- host launch ----------------
extern "C" void kernel_launch(void* const* d_in, const int* in_sizes, int n_in,
                              void* d_out, int out_size, void* d_ws,
                              size_t ws_size, hipStream_t stream) {
  (void)in_sizes; (void)n_in; (void)out_size; (void)ws_size;
  const float* x     = (const float*)d_in[0];
  const float* ln1_g = (const float*)d_in[1];
  const float* ln1_b = (const float*)d_in[2];
  const float* ln2_g = (const float*)d_in[3];
  const float* ln2_b = (const float*)d_in[4];
  const float* qkv_w = (const float*)d_in[5];
  const float* qkv_b = (const float*)d_in[6];
  const float* proj_w = (const float*)d_in[7];
  const float* proj_b = (const float*)d_in[8];
  const float* ffn_w1 = (const float*)d_in[9];
  const float* ffn_b1 = (const float*)d_in[10];
  const float* ffn_w2 = (const float*)d_in[11];
  const float* ffn_b2 = (const float*)d_in[12];
  const float* lscale = (const float*)d_in[13];
  float* out = (float*)d_out;

  char* ws = (char*)d_ws;
  const size_t SZ_WIN = (size_t)MWIN * DIMC * sizeof(_Float16);  // 32 MB
  const size_t OFF_HWIN = 0;
  const size_t OFF_Q    = OFF_HWIN + SZ_WIN;
  const size_t OFF_K    = OFF_Q + SZ_WIN;
  const size_t OFF_V    = OFF_K + SZ_WIN;
  const size_t OFF_OWIN = OFF_V + SZ_WIN;
  const size_t OFF_X1   = OFF_OWIN + SZ_WIN;                     // f32 x1
  const size_t SZ_X1    = (size_t)MTOK * DIMC * sizeof(float);
  const size_t OFF_WQ   = OFF_X1 + SZ_X1;   // f16 transposed weights
  const size_t OFF_WP   = OFF_WQ + (size_t)DIMC * 3 * DIMC * 2;
  const size_t OFF_W1   = OFF_WP + (size_t)DIMC * DIMC * 2;
  const size_t OFF_W2   = OFF_W1 + (size_t)DIMC * FFW * 2;
  const size_t OFF_H2   = OFF_HWIN;  // hwin dead after qkv gemm
  const size_t OFF_F1   = OFF_Q;     // q/k/v/owin dead after proj

  _Float16* hwin = (_Float16*)(ws + OFF_HWIN);
  _Float16* qb   = (_Float16*)(ws + OFF_Q);
  _Float16* kb   = (_Float16*)(ws + OFF_K);
  _Float16* vt   = (_Float16*)(ws + OFF_V);
  _Float16* owin = (_Float16*)(ws + OFF_OWIN);
  float*    x1   = (float*)(ws + OFF_X1);
  _Float16* wq16 = (_Float16*)(ws + OFF_WQ);
  _Float16* wp16 = (_Float16*)(ws + OFF_WP);
  _Float16* w116 = (_Float16*)(ws + OFF_W1);
  _Float16* w216 = (_Float16*)(ws + OFF_W2);
  _Float16* h2   = (_Float16*)(ws + OFF_H2);
  _Float16* f1   = (_Float16*)(ws + OFF_F1);

  // one-shot f32->f16 weight conversion + transpose (L2-resident after)
  k_w2ht<<<(DIMC * 3 * DIMC + 255) / 256, 256, 0, stream>>>(qkv_w, wq16,
                                                            DIMC, 3 * DIMC);
  k_w2ht<<<(DIMC * DIMC + 255) / 256, 256, 0, stream>>>(proj_w, wp16,
                                                        DIMC, DIMC);
  k_w2ht<<<(DIMC * FFW + 255) / 256, 256, 0, stream>>>(ffn_w1, w116,
                                                       DIMC, FFW);
  k_w2ht<<<(FFW * DIMC + 255) / 256, 256, 0, stream>>>(ffn_w2, w216,
                                                       FFW, DIMC);

  k_ln1<<<MWIN / 8, 256, 0, stream>>>(x, ln1_g, ln1_b, hwin);
  k_qkv<<<dim3(12, MWIN / 128), 256, 0, stream>>>(hwin, wq16, qkv_b,
                                                  qb, kb, vt);
  k_attn<<<BATCH * NWIN * NH, 128, 0, stream>>>(qb, kb, vt, owin);
  k_proj<<<dim3(4, MWIN / 128), 256, 0, stream>>>(owin, wp16, proj_b, x, x1);
  k_ln2<<<MTOK / 8, 256, 0, stream>>>(x1, ln2_g, ln2_b, h2);
  k_ffn1<<<dim3(16, MTOK / 128), 256, 0, stream>>>(h2, w116, ffn_b1, f1);
  k_ffn2<<<dim3(4, MTOK / 128), 256, 0, stream>>>(f1, w216, ffn_b2, lscale,
                                                  x1, out);
}
